// ModelFeatureNetwork_1494648619023
// MI455X (gfx1250) — compile-verified, matrix operand order checked by reading, not run on hardware
//
#include <hip/hip_runtime.h>

#define DEV __device__ __forceinline__

typedef __attribute__((ext_vector_type(16))) __bf16 v16bf;
typedef __attribute__((ext_vector_type(8)))  __bf16 bf16x8;
typedef __attribute__((ext_vector_type(8)))  float  v8f;

union Frag16 { v16bf v; bf16x8 h[2]; };

DEV float leakyf(float x) { return x > 0.f ? x : 0.2f * x; }

// pack two fp32 -> packed bf16 (truncation) with a single v_perm_b32:
// result = { hi[31:16], lo[31:16] }
DEV unsigned pack2(float lo, float hi) {
    return __builtin_amdgcn_perm(__float_as_uint(hi), __float_as_uint(lo), 0x07060302u);
}

// ---------------------------------------------------------------------------
// Generic WMMA GEMM: C[M,N] = act( A[M,K] @ W[K,N] + bias )
// fp32 global -> bf16 regs (v_perm pack) -> LDS (double buffered)
//   -> v_wmma_f32_16x16x32_bf16.
// Block tile 128x64, BK=32, 256 threads = 8 waves. Wave w owns a 32x32 output
// (rows 32*(w>>1), cols 32*(w&1)): 4 accumulators from 2 A frags x 2 B frags
// -> 2 ds_load_b128 per WMMA. All addressing is 32-bit; staging is branch-free
// (uniform fast path, clamp+select only on the rare K tail).
// ---------------------------------------------------------------------------
#define BM 128
#define BN 64
#define BK 32
#define LDA 40   // padded LDS row stride (elements); 80B, 16B-aligned
#define LDB 40

__global__ __launch_bounds__(256)
void wmma_gemm(const float* __restrict__ A, const float* __restrict__ W,
               const float* __restrict__ bias, float* __restrict__ C,
               int M, int N, int K, int actOut)
{
    __shared__ __align__(16) unsigned short sA[2][BM * LDA];  // [row][k] bf16 bits
    __shared__ __align__(16) unsigned short sB[2][BN * LDB];  // [col][k] bf16 bits

    const int tid  = threadIdx.x;
    const int wave = tid >> 5;
    const int lane = tid & 31;

    const int mBlock = blockIdx.y * BM;
    const int nBlock = blockIdx.x * BN;

    const int rgBase = (wave >> 1) * 32;  // wave's row group: 0,32,64,96
    const int cgBase = (wave & 1) * 32;   // wave's col group: 0,32

    v8f acc00 = {}, acc01 = {}, acc10 = {}, acc11 = {};

    // A staging: 128 rows x 32 k; thread owns rows (tid>>2) and 64+(tid>>2),
    // 8 consecutive k each.
    const int aRow = tid >> 2;
    const int aK   = (tid & 3) * 8;
    const int aGm0 = (mBlock + aRow      < M) ? (mBlock + aRow)      : (M - 1);
    const int aGm1 = (mBlock + 64 + aRow < M) ? (mBlock + 64 + aRow) : (M - 1);
    // B staging: 64 cols x 32 k; thread owns one col, 8 consecutive k.
    const int bNl  = tid & 63;
    const int bKb  = (tid >> 6) * 8;
    const int bGn  = nBlock + bNl;
    const int bGnc = (bGn < N) ? bGn : (N - 1);
    const bool colOK = (bGn < N);

    float aLo[8], aHi[8], bReg[8];

    const int nKt = (K + BK - 1) / BK;

    auto loadTile = [&](int kt) {
        const int k0 = kt * BK;
        if (k0 + BK <= K) {                       // uniform fast path
            const float4* a0 = (const float4*)(A + (unsigned)(aGm0 * K + k0 + aK));
            const float4* a1 = (const float4*)(A + (unsigned)(aGm1 * K + k0 + aK));
            float4 f0 = a0[0], f1 = a0[1], f2 = a1[0], f3 = a1[1];
            aLo[0]=f0.x; aLo[1]=f0.y; aLo[2]=f0.z; aLo[3]=f0.w;
            aLo[4]=f1.x; aLo[5]=f1.y; aLo[6]=f1.z; aLo[7]=f1.w;
            aHi[0]=f2.x; aHi[1]=f2.y; aHi[2]=f2.z; aHi[3]=f2.w;
            aHi[4]=f3.x; aHi[5]=f3.y; aHi[6]=f3.z; aHi[7]=f3.w;
            const int wOff = (k0 + bKb) * N + bGnc;   // 32-bit offsets
            #pragma unroll
            for (int j = 0; j < 8; ++j) {
                float v = W[(unsigned)(wOff + j * N)];
                bReg[j] = colOK ? v : 0.f;
            }
        } else {                                  // K tail (rare): clamp + select
            #pragma unroll
            for (int j = 0; j < 8; ++j) {
                int gk  = k0 + aK + j;
                int gkc = (gk < K) ? gk : (K - 1);
                float v0 = A[(unsigned)(aGm0 * K + gkc)];
                float v1 = A[(unsigned)(aGm1 * K + gkc)];
                aLo[j] = (gk < K) ? v0 : 0.f;
                aHi[j] = (gk < K) ? v1 : 0.f;
            }
            #pragma unroll
            for (int j = 0; j < 8; ++j) {
                int gk  = k0 + bKb + j;
                int gkc = (gk < K) ? gk : (K - 1);
                float v = W[(unsigned)(gkc * N + bGnc)];
                bReg[j] = (gk < K && colOK) ? v : 0.f;
            }
        }
    };
    auto storeTile = [&](int buf) {
        uint4 p0, p1, pb;
        p0.x = pack2(aLo[0], aLo[1]); p0.y = pack2(aLo[2], aLo[3]);
        p0.z = pack2(aLo[4], aLo[5]); p0.w = pack2(aLo[6], aLo[7]);
        p1.x = pack2(aHi[0], aHi[1]); p1.y = pack2(aHi[2], aHi[3]);
        p1.z = pack2(aHi[4], aHi[5]); p1.w = pack2(aHi[6], aHi[7]);
        pb.x = pack2(bReg[0], bReg[1]); pb.y = pack2(bReg[2], bReg[3]);
        pb.z = pack2(bReg[4], bReg[5]); pb.w = pack2(bReg[6], bReg[7]);
        *(uint4*)&sA[buf][aRow * LDA + aK]        = p0;
        *(uint4*)&sA[buf][(64 + aRow) * LDA + aK] = p1;
        *(uint4*)&sB[buf][bNl * LDB + bKb]        = pb;
    };

    loadTile(0);
    storeTile(0);
    __syncthreads();

    const int lrow = lane & 15;
    const int kOff = (lane < 16) ? 0 : 8;   // ISA 16-bit A/B fragment layout

    int buf = 0;
    for (int kt = 0; kt < nKt; ++kt) {
        const bool more = (kt + 1 < nKt);
        if (more) loadTile(kt + 1);          // global loads in flight during WMMA

        Frag16 a0, a1, b0, b1;
        a0.h[0] = *(const bf16x8*)&sA[buf][(rgBase + lrow) * LDA + kOff];
        a0.h[1] = *(const bf16x8*)&sA[buf][(rgBase + lrow) * LDA + 16 + kOff];
        a1.h[0] = *(const bf16x8*)&sA[buf][(rgBase + 16 + lrow) * LDA + kOff];
        a1.h[1] = *(const bf16x8*)&sA[buf][(rgBase + 16 + lrow) * LDA + 16 + kOff];
        b0.h[0] = *(const bf16x8*)&sB[buf][(cgBase + lrow) * LDB + kOff];
        b0.h[1] = *(const bf16x8*)&sB[buf][(cgBase + lrow) * LDB + 16 + kOff];
        b1.h[0] = *(const bf16x8*)&sB[buf][(cgBase + 16 + lrow) * LDB + kOff];
        b1.h[1] = *(const bf16x8*)&sB[buf][(cgBase + 16 + lrow) * LDB + 16 + kOff];

        acc00 = __builtin_amdgcn_wmma_f32_16x16x32_bf16(
                    false, a0.v, false, b0.v, (short)0, acc00, false, false);
        acc01 = __builtin_amdgcn_wmma_f32_16x16x32_bf16(
                    false, a0.v, false, b1.v, (short)0, acc01, false, false);
        acc10 = __builtin_amdgcn_wmma_f32_16x16x32_bf16(
                    false, a1.v, false, b0.v, (short)0, acc10, false, false);
        acc11 = __builtin_amdgcn_wmma_f32_16x16x32_bf16(
                    false, a1.v, false, b1.v, (short)0, acc11, false, false);

        if (more) storeTile(buf ^ 1);        // prefetch wait lands after the WMMAs
        __syncthreads();
        buf ^= 1;
    }

    // C/D layout: lanes 0-15 col=lane rows M=r; lanes 16-31 col=lane-16 rows M=8+r
    const int colLocal = lane & 15;
    const int rowOff   = (lane < 16) ? 0 : 8;
    #pragma unroll
    for (int r = 0; r < 8; ++r) {
        const int gm0 = mBlock + rgBase + rowOff + r;
        const int gm1 = gm0 + 16;
        const int gn0 = nBlock + cgBase + colLocal;
        const int gn1 = gn0 + 16;
        float b0v = 0.f, b1v = 0.f;
        if (bias) {
            b0v = (gn0 < N) ? bias[gn0] : 0.f;
            b1v = (gn1 < N) ? bias[gn1] : 0.f;
        }
        if (gm0 < M) {
            if (gn0 < N) {
                float v = acc00[r] + b0v; if (actOut) v = leakyf(v);
                C[(unsigned)(gm0 * N + gn0)] = v;
            }
            if (gn1 < N) {
                float v = acc01[r] + b1v; if (actOut) v = leakyf(v);
                C[(unsigned)(gm0 * N + gn1)] = v;
            }
        }
        if (gm1 < M) {
            if (gn0 < N) {
                float v = acc10[r] + b0v; if (actOut) v = leakyf(v);
                C[(unsigned)(gm1 * N + gn0)] = v;
            }
            if (gn1 < N) {
                float v = acc11[r] + b1v; if (actOut) v = leakyf(v);
                C[(unsigned)(gm1 * N + gn1)] = v;
            }
        }
    }
}

// ---------------------------------------------------------------------------
// Tiny GEMM for the 8-row drug branch: one thread per output element.
// ---------------------------------------------------------------------------
__global__ void small_gemm(const float* __restrict__ A, const float* __restrict__ W,
                           const float* __restrict__ bias, float* __restrict__ C,
                           int R, int K, int N, int act)
{
    int idx = blockIdx.x * blockDim.x + threadIdx.x;
    if (idx >= R * N) return;
    int r = idx / N, c = idx % N;
    float s = bias ? bias[c] : 0.f;
    const float* a  = A + (unsigned)(r * K);
    const float* wp = W + c;
    for (int k = 0; k < K; ++k) { s += a[k] * wp[0]; wp += N; }
    C[idx] = act ? leakyf(s) : s;
}

// out[r, 0:400] = leaky( [X0[r,0:200], X1[r,0:200]] )
__global__ void concat_leaky(const float* __restrict__ X0, const float* __restrict__ X1,
                             float* __restrict__ out, int rows)
{
    int idx = blockIdx.x * blockDim.x + threadIdx.x;
    if (idx >= rows * 400) return;
    int r = idx / 400, c = idx % 400;
    float v = (c < 200) ? X0[(unsigned)(r * 200 + c)] : X1[(unsigned)(r * 200 + (c - 200))];
    out[idx] = leakyf(v);
}

__global__ void deg_init(float* deg, int n)
{
    int i = blockIdx.x * blockDim.x + threadIdx.x;
    if (i < n) deg[i] = 2.0f;   // improved GCN: A + 2I
}

__global__ void deg_accum(const int* __restrict__ dst, float* deg, int E)
{
    int e = blockIdx.x * blockDim.x + threadIdx.x;
    if (e < E) atomicAdd(&deg[dst[e]], 1.0f);
}

// out[i,c] = (2/deg_i) * h[i,c] + bias[c]   (self loops, weight 2)
__global__ void gcn_self(const float* __restrict__ h, const float* __restrict__ deg,
                         const float* __restrict__ bias, float* __restrict__ out,
                         int n, int d)
{
    int idx = blockIdx.x * blockDim.x + threadIdx.x;
    if (idx >= n * d) return;
    int i = idx / d, c = idx % d;
    out[idx] = (2.0f / deg[i]) * h[idx] + bias[c];
}

// out[dst] += rsqrt(deg[src])*rsqrt(deg[dst]) * h[src]; 8 channels per thread
__global__ void gcn_scatter(const float* __restrict__ h, const float* __restrict__ deg,
                            const int* __restrict__ src, const int* __restrict__ dst,
                            float* __restrict__ out, int E, int d)
{
    const int CH = 8;
    const int chunks = d / CH;              // 200/8 = 25
    int idx = blockIdx.x * blockDim.x + threadIdx.x;
    if (idx >= E * chunks) return;
    int e  = idx / chunks;
    int c0 = (idx % chunks) * CH;
    int s = src[e], t = dst[e];
    float norm = rsqrtf(deg[s]) * rsqrtf(deg[t]);
    const float* hp = h   + (unsigned)(s * d + c0);
    float*       op = out + (unsigned)(t * d + c0);
    #pragma unroll
    for (int j = 0; j < CH; ++j) atomicAdd(&op[j], norm * hp[j]);
}

// sigmoid(cosine(d[row/P], px[row])), one wave (32 lanes) per row
__global__ void cosine_out(const float* __restrict__ dmat, const float* __restrict__ px,
                           float* __restrict__ out, int P, int d, int total)
{
    int row  = blockIdx.x * (blockDim.x >> 5) + (threadIdx.x >> 5);
    int lane = threadIdx.x & 31;
    if (row >= total) return;                 // wave-uniform exit
    int b = row / P;
    const float* dv = dmat + (unsigned)(b * d);
    const float* pv = px   + (unsigned)(row * d);
    float num = 0.f, nd = 0.f, np = 0.f;
    for (int k = lane; k < d; k += 32) {
        float a = dv[k], p = pv[k];
        num += a * p; nd += a * a; np += p * p;
    }
    #pragma unroll
    for (int m = 16; m; m >>= 1) {
        num += __shfl_xor(num, m, 32);
        nd  += __shfl_xor(nd,  m, 32);
        np  += __shfl_xor(np,  m, 32);
    }
    if (lane == 0) {
        float den = fmaxf(sqrtf(nd), 1e-8f) * fmaxf(sqrtf(np), 1e-8f);
        out[row] = 1.0f / (1.0f + expf(-num / den));
    }
}

// ---------------------------------------------------------------------------
extern "C" void kernel_launch(void* const* d_in, const int* in_sizes, int n_in,
                              void* d_out, int out_size, void* d_ws, size_t ws_size,
                              hipStream_t stream)
{
    const float* PPI_x    = (const float*)d_in[0];   // [8000,1024]
    const float* protmol  = (const float*)d_in[1];   // [8000,8192]
    const float* drugf    = (const float*)d_in[2];   // [8,1024]
    const float* drugmol  = (const float*)d_in[3];   // [8,1024]
    const int*   eidx     = (const int*)  d_in[4];   // [2,512000]
    const float* hpo_drug_W = (const float*)d_in[5];
    const float* hpo_drug_b = (const float*)d_in[6];
    const float* hpo_prot_W = (const float*)d_in[7];
    const float* hpo_prot_b = (const float*)d_in[8];
    const float* mp_W1 = (const float*)d_in[9];
    const float* mp_b1 = (const float*)d_in[10];
    const float* mp_W2 = (const float*)d_in[11];
    const float* mp_b2 = (const float*)d_in[12];
    const float* md_W1 = (const float*)d_in[13];
    const float* md_b1 = (const float*)d_in[14];
    const float* md_W2 = (const float*)d_in[15];
    const float* md_b2 = (const float*)d_in[16];
    const float* pl1_W = (const float*)d_in[17];
    const float* pl1_b = (const float*)d_in[18];
    const float* dl1_W = (const float*)d_in[19];
    const float* dl1_b = (const float*)d_in[20];
    const float* dl2_W = (const float*)d_in[21];
    const float* dl2_b = (const float*)d_in[22];
    const float* g1_W  = (const float*)d_in[23];
    const float* g1_b  = (const float*)d_in[24];
    const float* g2_W  = (const float*)d_in[25];
    const float* g2_b  = (const float*)d_in[26];

    float* out = (float*)d_out;
    float* ws  = (float*)d_ws;

    const int Nn = 8000, Bsz = 8, P = 1000, E = 512000, D = 200;

    // workspace layout (floats), reuse-aware (~32 MB total)
    size_t o = 0;
    float* bufA   = ws + o; o += (size_t)Nn * 200;   // px -> px2 -> h2
    float* bufB   = ws + o; o += (size_t)Nn * 200;   // pm -> h1 -> gout2
    float* bufBig = ws + o; o += (size_t)Nn * 400;   // H1[8000,256] then cat[8000,400]
    float* bufC   = ws + o; o += (size_t)Nn * 200;   // gout1
    float* deg    = ws + o; o += 8192;
    float* ddf    = ws + o; o += Bsz * 200;
    float* ddh    = ws + o; o += Bsz * 256;
    float* ddm    = ws + o; o += Bsz * 200;
    float* dcat   = ws + o; o += Bsz * 400;
    float* dd1    = ws + o; o += Bsz * 200;
    float* dvec   = ws + o; o += Bsz * 200;
    (void)ws_size; (void)n_in; (void)in_sizes; (void)out_size;

    auto cdiv = [](int a, int b) { return (a + b - 1) / b; };
    const int T = 256;

    // ---- drug branch (tiny) ----
    small_gemm<<<cdiv(Bsz * 200, T), T, 0, stream>>>(drugf,   hpo_drug_W, hpo_drug_b, ddf, Bsz, 1024, 200, 0);
    small_gemm<<<cdiv(Bsz * 256, T), T, 0, stream>>>(drugmol, md_W1,      md_b1,      ddh, Bsz, 1024, 256, 1);
    small_gemm<<<cdiv(Bsz * 200, T), T, 0, stream>>>(ddh,     md_W2,      md_b2,      ddm, Bsz,  256, 200, 0);
    concat_leaky<<<cdiv(Bsz * 400, T), T, 0, stream>>>(ddm, ddf, dcat, Bsz);
    small_gemm<<<cdiv(Bsz * 200, T), T, 0, stream>>>(dcat, dl1_W, dl1_b, dd1,  Bsz, 400, 200, 1);
    small_gemm<<<cdiv(Bsz * 200, T), T, 0, stream>>>(dd1,  dl2_W, dl2_b, dvec, Bsz, 200, 200, 0);

    // ---- protein branch (WMMA) ----
    dim3 gPX (cdiv(200, BN), cdiv(Nn, BM));   // (4, 63)
    dim3 gH1 (cdiv(256, BN), cdiv(Nn, BM));   // (4, 63)
    wmma_gemm<<<gPX, T, 0, stream>>>(PPI_x,   hpo_prot_W, hpo_prot_b, bufA,   Nn, 200, 1024, 0); // px
    wmma_gemm<<<gH1, T, 0, stream>>>(protmol, mp_W1,      mp_b1,      bufBig, Nn, 256, 8192, 1); // H1
    wmma_gemm<<<gPX, T, 0, stream>>>(bufBig,  mp_W2,      mp_b2,      bufB,   Nn, 200,  256, 0); // pm
    concat_leaky<<<cdiv(Nn * 400, T), T, 0, stream>>>(bufA, bufB, bufBig, Nn);                    // cat
    wmma_gemm<<<gPX, T, 0, stream>>>(bufBig,  pl1_W,      pl1_b,      bufA,   Nn, 200,  400, 0); // px2

    // ---- GCN degrees ----
    deg_init <<<cdiv(Nn, T), T, 0, stream>>>(deg, Nn);
    deg_accum<<<cdiv(E,  T), T, 0, stream>>>(eidx + E, deg, E);

    // ---- GCN layer 1 ----
    wmma_gemm  <<<gPX, T, 0, stream>>>(bufA, g1_W, nullptr, bufB, Nn, 200, 200, 0);              // h1
    gcn_self   <<<cdiv(Nn * D, T), T, 0, stream>>>(bufB, deg, g1_b, bufC, Nn, D);
    gcn_scatter<<<cdiv(E * 25, T), T, 0, stream>>>(bufB, deg, eidx, eidx + E, bufC, E, D);

    // ---- GCN layer 2 ----
    wmma_gemm  <<<gPX, T, 0, stream>>>(bufC, g2_W, nullptr, bufA, Nn, 200, 200, 0);              // h2
    gcn_self   <<<cdiv(Nn * D, T), T, 0, stream>>>(bufA, deg, g2_b, bufB, Nn, D);
    gcn_scatter<<<cdiv(E * 25, T), T, 0, stream>>>(bufA, deg, eidx, eidx + E, bufB, E, D);

    // ---- cosine similarity + sigmoid ----
    cosine_out<<<cdiv(Nn, 8), T, 0, stream>>>(dvec, bufB, out, P, D, Nn);
}
